// EGNNLayer_781684048540
// MI455X (gfx1250) — matrix operand order repaired and checked by reading, not compiled, not run on hardware
//
#include <hip/hip_runtime.h>
#include <hip/hip_bf16.h>
#include <stdint.h>

#define HID 128
#define NNODES 50000
#define NEDGES 640000

typedef __attribute__((ext_vector_type(16))) __bf16 v16bf;
typedef __attribute__((ext_vector_type(8)))  float  v8f;

union Frag16 { v16bf v; uint32_t u[8]; uint4 q[2]; };

// pack two f32 -> two bf16 (round-half-up): used only in one-time prep kernels
__device__ __forceinline__ uint32_t pack_bf16(float a, float b) {
  uint32_t ua = __float_as_uint(a) + 0x8000u;
  uint32_t ub = __float_as_uint(b) + 0x8000u;
  return __builtin_amdgcn_perm(ub, ua, 0x07060302u);  // [ub.b3,ub.b2,ua.b3,ua.b2]
}

// truncating f32 -> bf16 (hot path: 1 VALU + ds_store_b16)
__device__ __forceinline__ uint16_t f32_to_bf16(float a) {
  return (uint16_t)(__float_as_uint(a) >> 16);
}

// silu with hardware rcp: v_exp + v_add + v_rcp + v_mul
__device__ __forceinline__ float silu_f(float v) {
  return v * __builtin_amdgcn_rcpf(1.0f + __expf(-v));
}

__device__ __forceinline__ float tanh_f(float v) {
  return 1.0f - 2.0f * __builtin_amdgcn_rcpf(__expf(2.0f * v) + 1.0f);
}

__device__ __forceinline__ v8f vzero8() {
  v8f z;
#pragma unroll
  for (int i = 0; i < 8; ++i) z[i] = 0.0f;
  return z;
}

__device__ __forceinline__ v8f wmma_bf16(const Frag16& a, const Frag16& b, v8f c) {
  // D = A(16x32 bf16) * B(32x16 bf16) + C(16x16 f32)
  return __builtin_amdgcn_wmma_f32_16x16x32_bf16(false, a.v, false, b.v,
                                                 (short)0, c, false, false);
}

// A fragment from a pre-packed bf16 row (row = 64 dwords = 128 bf16).
// Per-lane K pairs {0,2,...,14}+16g -> dwords {0..7}+8g.
__device__ __forceinline__ void load_a_bf16(Frag16& a, const uint32_t* rowp, int kt, int g) {
  a.q[0] = *(const uint4*)(rowp + kt * 16 + 8 * g);
  a.q[1] = *(const uint4*)(rowp + kt * 16 + 4 + 8 * g);
}

// Load all 8 B fragments of one K-step into distinct registers (constant offsets),
// so the compiler can clause the 16 b128 loads and use partial s_wait_loadcnt.
__device__ __forceinline__ void load_b8(Frag16 b[8], const uint32_t* bbase, int kt) {
#pragma unroll
  for (int nt = 0; nt < 8; ++nt) {
    b[nt].q[0] = *(const uint4*)(bbase + kt * 2048 + nt * 128);
    b[nt].q[1] = *(const uint4*)(bbase + kt * 2048 + nt * 128 + 4);
  }
}

// ---- K=128 GEMM from LDS bf16 activations: acc[nt] += act(16x128) * B ----
__device__ __forceinline__ void gemm128_lds(const uint16_t* sact, int erow, int g, int n,
                                            const uint32_t* __restrict__ B, v8f acc[8]) {
  const uint32_t* bbase = B + (size_t)((g * HID + n) * 8);
  const uint16_t* abase = &sact[(erow + n) * HID + 16 * g];
#pragma unroll
  for (int kt = 0; kt < 4; ++kt) {
    Frag16 a;
    a.q[0] = *(const uint4*)(abase + kt * 32);
    a.q[1] = *(const uint4*)(abase + kt * 32 + 8);
    Frag16 b[8];
    load_b8(b, bbase, kt);
#pragma unroll
    for (int nt = 0; nt < 8; ++nt) acc[nt] = wmma_bf16(a, b[nt], acc[nt]);
  }
}

// ---- init: zero m_aggr, x_out = x ----
__global__ void egnn_init_kernel(const float* __restrict__ x,
                                 float* __restrict__ m_aggr,
                                 float* __restrict__ x_out) {
  int stride = gridDim.x * blockDim.x;
  for (int i = blockIdx.x * blockDim.x + threadIdx.x; i < NNODES * HID; i += stride) {
    m_aggr[i] = 0.0f;
    if (i < NNODES * 3) x_out[i] = x[i];
  }
}

// ---- f32 -> packed bf16 pairs (for h and m_aggr activations) ----
__global__ void cvt_bf16_kernel(const float* __restrict__ in, uint32_t* __restrict__ out,
                                int npairs) {
  int stride = gridDim.x * blockDim.x;
  for (int i = blockIdx.x * blockDim.x + threadIdx.x; i < npairs; i += stride) {
    float2 t = *(const float2*)(in + 2 * i);
    out[i] = pack_bf16(t.x, t.y);
  }
}

// ---- pack W[out=N, in=K(rowStride)] -> bf16 B-fragment layout ----
// dword i = ((ktile*2+g)*N + n)*8 + v holds {W[n][k0], W[n][k0+1]}, k0 = ktile*32+g*16+2v
__global__ void pack_b_kernel(const float* __restrict__ W, uint32_t* __restrict__ out,
                              int K, int N, int rowStride) {
  int total = (K / 2) * N;
  int stride = gridDim.x * blockDim.x;
  for (int i = blockIdx.x * blockDim.x + threadIdx.x; i < total; i += stride) {
    int v = i & 7;
    int rest = i >> 3;
    int n = rest % N; rest /= N;
    int g = rest & 1;
    int ktile = rest >> 1;
    int k0 = ktile * 32 + g * 16 + 2 * v;
    out[i] = pack_bf16(W[n * rowStride + k0], W[n * rowStride + k0 + 1]);
  }
}

// ---- edge kernel: 4 waves x 16 edges per block ----
__global__ __launch_bounds__(128) void egnn_edge_kernel(
    const float* __restrict__ x, const uint32_t* __restrict__ hb,
    const unsigned long long* __restrict__ eidx,
    const uint32_t* __restrict__ B1, const float* __restrict__ ew1, const float* __restrict__ eb1,
    const uint32_t* __restrict__ B2, const float* __restrict__ eb2,
    const uint32_t* __restrict__ B3, const float* __restrict__ eb3,
    const uint32_t* __restrict__ Bc1, const float* __restrict__ cb1,
    const float* __restrict__ cw2, const float* __restrict__ cb2,
    float* __restrict__ m_aggr, float* __restrict__ x_out) {
  __shared__ uint16_t sact[64 * HID];
  __shared__ int   ssrc[64];
  __shared__ int   sdst[64];
  __shared__ float sd2[64];
  __shared__ float srel[64][3];

  const int tid = threadIdx.x;
  const int lane = tid & 31;
  const int wv = tid >> 5;
  const int g = lane >> 4;
  const int n = lane & 15;
  const int erow = wv * 16;

  if (tid < 64) {
    int e = blockIdx.x * 64 + tid;
    int s = (int)eidx[e];
    int d = (int)eidx[NEDGES + e];
    float rx = x[s * 3 + 0] - x[d * 3 + 0];
    float ry = x[s * 3 + 1] - x[d * 3 + 1];
    float rz = x[s * 3 + 2] - x[d * 3 + 2];
    ssrc[tid] = s; sdst[tid] = d;
    srel[tid][0] = rx; srel[tid][1] = ry; srel[tid][2] = rz;
    sd2[tid] = rx * rx + ry * ry + rz * rz;
  }
  __syncthreads();

  // ---- layer 1: [h[src], h[dst]] (K=256 via WMMA) + d2 rank-1 epilogue ----
  v8f acc[8];
#pragma unroll
  for (int nt = 0; nt < 8; ++nt) acc[nt] = vzero8();

  {
    const uint32_t* srow = hb + (size_t)ssrc[erow + n] * 64;
    const uint32_t* drow = hb + (size_t)sdst[erow + n] * 64;
    const uint32_t* bbase = B1 + (size_t)((g * HID + n) * 8);
#pragma unroll
    for (int kt = 0; kt < 8; ++kt) {
      Frag16 a;
      load_a_bf16(a, (kt < 4) ? srow : drow, kt & 3, g);
      Frag16 b[8];
      load_b8(b, bbase, kt);
#pragma unroll
      for (int nt = 0; nt < 8; ++nt) acc[nt] = wmma_bf16(a, b[nt], acc[nt]);
    }
  }
  {
    float bias1[8], wl1[8];
#pragma unroll
    for (int nt = 0; nt < 8; ++nt) {
      bias1[nt] = eb1[nt * 16 + n];
      wl1[nt]   = ew1[(nt * 16 + n) * 257 + 256];  // d2 column of ew1
    }
#pragma unroll
    for (int r = 0; r < 8; ++r) {
      int m = r + 8 * g;
      float d2v = sd2[erow + m];
      uint16_t* srow = &sact[(erow + m) * HID + n];
#pragma unroll
      for (int nt = 0; nt < 8; ++nt)
        srow[nt * 16] = f32_to_bf16(silu_f(acc[nt][r] + bias1[nt] + d2v * wl1[nt]));
    }
  }
  __syncthreads();

  // ---- layer 2 ----
  v8f acc2[8];
#pragma unroll
  for (int nt = 0; nt < 8; ++nt) acc2[nt] = vzero8();
  gemm128_lds(sact, erow, g, n, B2, acc2);
  {
    float bias2[8];
#pragma unroll
    for (int nt = 0; nt < 8; ++nt) bias2[nt] = eb2[nt * 16 + n];
#pragma unroll
    for (int r = 0; r < 8; ++r) {
      uint16_t* srow = &sact[(erow + r + 8 * g) * HID + n];
#pragma unroll
      for (int nt = 0; nt < 8; ++nt)
        srow[nt * 16] = f32_to_bf16(silu_f(acc2[nt][r] + bias2[nt]));
    }
  }
  __syncthreads();

  // ---- layer 3: m_ij -> atomics + LDS ----
  v8f acc3[8];
#pragma unroll
  for (int nt = 0; nt < 8; ++nt) acc3[nt] = vzero8();
  gemm128_lds(sact, erow, g, n, B3, acc3);
  __syncthreads();
  {
    float bias3[8];
#pragma unroll
    for (int nt = 0; nt < 8; ++nt) bias3[nt] = eb3[nt * 16 + n];
#pragma unroll
    for (int r = 0; r < 8; ++r) {
      int m = r + 8 * g;
      float* abase = &m_aggr[(size_t)ssrc[erow + m] * HID + n];
      uint16_t* srow = &sact[(erow + m) * HID + n];
#pragma unroll
      for (int nt = 0; nt < 8; ++nt) {
        float val = acc3[nt][r] + bias3[nt];
        atomicAdd(abase + nt * 16, val);        // segment_sum(m_ij, src)
        srow[nt * 16] = f32_to_bf16(val);
      }
    }
  }
  __syncthreads();

  // ---- coord path: silu(m_ij @ cw1^T + cb1) . cw2 -> tanh -> delta_x ----
  v8f accc[8];
#pragma unroll
  for (int nt = 0; nt < 8; ++nt) accc[nt] = vzero8();
  gemm128_lds(sact, erow, g, n, Bc1, accc);

  float biasc[8], w2c[8];
#pragma unroll
  for (int nt = 0; nt < 8; ++nt) {
    biasc[nt] = cb1[nt * 16 + n];
    w2c[nt]   = cw2[nt * 16 + n];
  }
  float cb2v = cb2[0];
#pragma unroll
  for (int r = 0; r < 8; ++r) {
    float s = 0.0f;
#pragma unroll
    for (int nt = 0; nt < 8; ++nt) s += silu_f(accc[nt][r] + biasc[nt]) * w2c[nt];
    s += __shfl_xor(s, 1);
    s += __shfl_xor(s, 2);
    s += __shfl_xor(s, 4);
    s += __shfl_xor(s, 8);
    if (n == 0) {
      int e = erow + r + 8 * g;
      float coef = tanh_f(s + cb2v);
      float* xb = &x_out[(size_t)ssrc[e] * 3];
      atomicAdd(xb + 0, srel[e][0] * coef);
      atomicAdd(xb + 1, srel[e][1] * coef);
      atomicAdd(xb + 2, srel[e][2] * coef);
    }
  }
}

// ---- node kernel: h_out = h + mlp3([h, m_aggr]) ----
__global__ __launch_bounds__(128) void egnn_node_kernel(
    const float* __restrict__ h, const uint32_t* __restrict__ hb,
    const uint32_t* __restrict__ mb,
    const uint32_t* __restrict__ Bn1, const float* __restrict__ nb1,
    const uint32_t* __restrict__ Bn2, const float* __restrict__ nb2,
    const uint32_t* __restrict__ Bn3, const float* __restrict__ nb3,
    float* __restrict__ h_out) {
  __shared__ uint16_t sact[64 * HID];
  const int tid = threadIdx.x;
  const int lane = tid & 31;
  const int wv = tid >> 5;
  const int g = lane >> 4;
  const int n = lane & 15;
  const int erow = wv * 16;
  const int rowbase = blockIdx.x * 64 + wv * 16;

  v8f acc[8];
#pragma unroll
  for (int nt = 0; nt < 8; ++nt) acc[nt] = vzero8();

  const int node = min(rowbase + n, NNODES - 1);  // clamp tail reads; writes masked below
  {
    const uint32_t* hrow = hb + (size_t)node * 64;
    const uint32_t* mrow = mb + (size_t)node * 64;
    const uint32_t* bbase = Bn1 + (size_t)((g * HID + n) * 8);
#pragma unroll
    for (int kt = 0; kt < 8; ++kt) {
      Frag16 a;
      load_a_bf16(a, (kt < 4) ? hrow : mrow, kt & 3, g);
      Frag16 b[8];
      load_b8(b, bbase, kt);
#pragma unroll
      for (int nt = 0; nt < 8; ++nt) acc[nt] = wmma_bf16(a, b[nt], acc[nt]);
    }
  }
  {
    float bias1[8];
#pragma unroll
    for (int nt = 0; nt < 8; ++nt) bias1[nt] = nb1[nt * 16 + n];
#pragma unroll
    for (int r = 0; r < 8; ++r) {
      uint16_t* srow = &sact[(erow + r + 8 * g) * HID + n];
#pragma unroll
      for (int nt = 0; nt < 8; ++nt)
        srow[nt * 16] = f32_to_bf16(silu_f(acc[nt][r] + bias1[nt]));
    }
  }
  __syncthreads();

  v8f acc2[8];
#pragma unroll
  for (int nt = 0; nt < 8; ++nt) acc2[nt] = vzero8();
  gemm128_lds(sact, erow, g, n, Bn2, acc2);
  {
    float bias2[8];
#pragma unroll
    for (int nt = 0; nt < 8; ++nt) bias2[nt] = nb2[nt * 16 + n];
#pragma unroll
    for (int r = 0; r < 8; ++r) {
      uint16_t* srow = &sact[(erow + r + 8 * g) * HID + n];
#pragma unroll
      for (int nt = 0; nt < 8; ++nt)
        srow[nt * 16] = f32_to_bf16(silu_f(acc2[nt][r] + bias2[nt]));
    }
  }
  __syncthreads();

  v8f acc3[8];
#pragma unroll
  for (int nt = 0; nt < 8; ++nt) acc3[nt] = vzero8();
  gemm128_lds(sact, erow, g, n, Bn3, acc3);
  {
    float bias3[8];
#pragma unroll
    for (int nt = 0; nt < 8; ++nt) bias3[nt] = nb3[nt * 16 + n];
#pragma unroll
    for (int r = 0; r < 8; ++r) {
      int nd = rowbase + r + 8 * g;
      if (nd < NNODES) {
        const float* hrow = &h[(size_t)nd * HID + n];
        float* orow = &h_out[(size_t)nd * HID + n];
#pragma unroll
        for (int nt = 0; nt < 8; ++nt)
          orow[nt * 16] = hrow[nt * 16] + acc3[nt][r] + bias3[nt];
      }
    }
  }
}

extern "C" void kernel_launch(void* const* d_in, const int* in_sizes, int n_in,
                              void* d_out, int out_size, void* d_ws, size_t ws_size,
                              hipStream_t stream) {
  const float* x   = (const float*)d_in[0];
  const float* h   = (const float*)d_in[1];
  const unsigned long long* eidx = (const unsigned long long*)d_in[2];
  const float* ew1 = (const float*)d_in[3];
  const float* eb1 = (const float*)d_in[4];
  const float* ew2 = (const float*)d_in[5];
  const float* eb2 = (const float*)d_in[6];
  const float* ew3 = (const float*)d_in[7];
  const float* eb3 = (const float*)d_in[8];
  const float* cw1 = (const float*)d_in[9];
  const float* cb1 = (const float*)d_in[10];
  const float* cw2 = (const float*)d_in[11];
  const float* cb2 = (const float*)d_in[12];
  const float* nw1 = (const float*)d_in[13];
  const float* nb1 = (const float*)d_in[14];
  const float* nw2 = (const float*)d_in[15];
  const float* nb2 = (const float*)d_in[16];
  const float* nw3 = (const float*)d_in[17];
  const float* nb3 = (const float*)d_in[18];

  float* x_out = (float*)d_out;                      // [N,3]
  float* h_out = (float*)d_out + (size_t)NNODES * 3; // [N,H]

  // workspace layout
  float* m_aggr = (float*)d_ws;                               // N*H f32
  uint32_t* wsb = (uint32_t*)((char*)d_ws + (size_t)NNODES * HID * sizeof(float));
  uint32_t* B1  = wsb;            // ew1: 16384 dwords
  uint32_t* B2  = B1 + 16384;     // 8192
  uint32_t* B3  = B2 + 8192;
  uint32_t* Bc1 = B3 + 8192;
  uint32_t* Bn1 = Bc1 + 8192;     // 16384
  uint32_t* Bn2 = Bn1 + 16384;
  uint32_t* Bn3 = Bn2 + 8192;
  uint32_t* hb  = Bn3 + 8192;                 // h as packed bf16: N*H/2 dwords
  uint32_t* mb  = hb + (size_t)NNODES * HID / 2;  // m_aggr as packed bf16

  egnn_init_kernel<<<2048, 256, 0, stream>>>(x, m_aggr, x_out);

  pack_b_kernel<<<128, 256, 0, stream>>>(ew1, B1, 256, HID, 257);
  pack_b_kernel<<<64, 256, 0, stream>>>(ew2, B2, 128, HID, 128);
  pack_b_kernel<<<64, 256, 0, stream>>>(ew3, B3, 128, HID, 128);
  pack_b_kernel<<<64, 256, 0, stream>>>(cw1, Bc1, 128, HID, 128);
  pack_b_kernel<<<128, 256, 0, stream>>>(nw1, Bn1, 256, HID, 256);
  pack_b_kernel<<<64, 256, 0, stream>>>(nw2, Bn2, 128, HID, 128);
  pack_b_kernel<<<64, 256, 0, stream>>>(nw3, Bn3, 128, HID, 128);

  cvt_bf16_kernel<<<2048, 256, 0, stream>>>(h, hb, NNODES * HID / 2);

  egnn_edge_kernel<<<NEDGES / 64, 128, 0, stream>>>(
      x, hb, eidx, B1, ew1, eb1, B2, eb2, B3, eb3, Bc1, cb1, cw2, cb2, m_aggr, x_out);

  cvt_bf16_kernel<<<2048, 256, 0, stream>>>(m_aggr, mb, NNODES * HID / 2);

  egnn_node_kernel<<<(NNODES + 63) / 64, 128, 0, stream>>>(
      h, hb, mb, Bn1, nb1, Bn2, nb2, Bn3, nb3, h_out);
}
